// SarvamMoEAttention_73847667687615
// MI455X (gfx1250) — matrix operand order repaired and checked by reading, not compiled, not run on hardware
//
#include <hip/hip_runtime.h>
#include <hip/hip_bf16.h>
#include <cstdint>

typedef __bf16 bf16_t;
typedef __attribute__((ext_vector_type(16))) __bf16 v16bf;
typedef __attribute__((ext_vector_type(8)))  __bf16 v8bf;
typedef __attribute__((ext_vector_type(8)))  float   v8f;
typedef __attribute__((ext_vector_type(4)))  int     v4i;

#define SEQ   2048
#define HID   2048
#define HD    128
#define NH    16
#define NKV   4
#define SA2   72      // LDS bf16 stride for 64-wide K staging tiles (144B rows, 16B aligned)
#define SQ    136     // LDS bf16 stride for attention tiles
#define SS    132     // LDS f32 stride for score / C tiles
#define TILE_ELEMS (128 * SA2)

// ---- async global->LDS path (gfx1250), with portable fallback --------------
#if __has_builtin(__builtin_amdgcn_global_load_async_to_lds_b128) && \
    __has_builtin(__builtin_amdgcn_s_wait_asynccnt)
#define USE_ASYNC 1
#else
#define USE_ASYNC 0
#endif

typedef __attribute__((address_space(1))) v4i* gptr_v4i;
typedef __attribute__((address_space(3))) v4i* lptr_v4i;

static __device__ __forceinline__ void async_copy_b128(bf16_t* lds, const bf16_t* g) {
#if USE_ASYNC
  __builtin_amdgcn_global_load_async_to_lds_b128(
      (gptr_v4i)const_cast<bf16_t*>(g), (lptr_v4i)lds, 0, 0);
#else
  *(v8bf*)lds = *(const v8bf*)g;
#endif
}
static __device__ __forceinline__ void async_wait0() {
#if USE_ASYNC
  __builtin_amdgcn_s_wait_asynccnt(0);
#endif
}
static __device__ __forceinline__ void async_wait8() {
#if USE_ASYNC
  __builtin_amdgcn_s_wait_asynccnt(8);
#endif
}

// ---------------------------------------------------------------------------
static __device__ __forceinline__ v8f vzero8() {
  v8f z = {0.f,0.f,0.f,0.f,0.f,0.f,0.f,0.f};
  return z;
}

static __device__ __forceinline__ v8f wmma_bf16(v16bf a, v16bf b, v8f c) {
  return __builtin_amdgcn_wmma_f32_16x16x32_bf16(false, a, false, b, (short)0, c, false, false);
}

// A fragment: 16x32 bf16 tile, row-major in LDS with given stride.
static __device__ __forceinline__ v16bf load_frag_a(const bf16_t* base, int stride, int lane) {
  union { v16bf v; v8bf h[2]; } u;
  const bf16_t* p = base + (lane & 15) * stride + ((lane & 16) ? 8 : 0);
  u.h[0] = *(const v8bf*)(p);
  u.h[1] = *(const v8bf*)(p + 16);
  return u.v;
}

// B fragment: 32x16 bf16 stored transposed ([N][K]) in LDS with given stride.
static __device__ __forceinline__ v16bf load_frag_b(const bf16_t* base, int stride, int lane) {
  union { v16bf v; v8bf h[2]; } u;
  const bf16_t* p = base + (lane & 15) * stride + ((lane & 16) ? 8 : 0);
  u.h[0] = *(const v8bf*)(p);
  u.h[1] = *(const v8bf*)(p + 16);
  return u.v;
}

// stage one 128x64 bf16 tile (global row stride HID) into LDS (stride SA2)
static __device__ __forceinline__ void stage_tile64(const bf16_t* g, bf16_t* lds, int t) {
  const int row = t >> 1, cb = (t & 1) * 32;        // 2 threads/row, 32 elems (64B) each
  const bf16_t* s = g + (size_t)row * HID + cb;
  bf16_t* d = lds + row * SA2 + cb;
#pragma unroll
  for (int i = 0; i < 4; ++i) async_copy_b128(d + i * 8, s + i * 8);
}

// ---------------------------------------------------------------------------
// Kernel 0: one-shot fp32 -> bf16 conversion
// ---------------------------------------------------------------------------
__global__ __launch_bounds__(256) void cvt_bf16_kernel(
    const float* __restrict__ src, bf16_t* __restrict__ dst, int n)
{
  const int i = (blockIdx.x * 256 + threadIdx.x) * 8;
  if (i < n) {
    const float4 a = ((const float4*)(src + i))[0];
    const float4 b = ((const float4*)(src + i))[1];
    v8bf o;
    o[0]=(bf16_t)a.x; o[1]=(bf16_t)a.y; o[2]=(bf16_t)a.z; o[3]=(bf16_t)a.w;
    o[4]=(bf16_t)b.x; o[5]=(bf16_t)b.y; o[6]=(bf16_t)b.z; o[7]=(bf16_t)b.w;
    ((v8bf*)(dst + i))[0] = o;
  }
}

// ---------------------------------------------------------------------------
// Kernel 1: QKV GEMM (C = H * Wqkv^T) on bf16, async double-buffered K=64,
// fused RMSNorm + RoPE + bf16 pack.  grid=(24,32), block=256 (4x2 waves).
// ---------------------------------------------------------------------------
__global__ __launch_bounds__(256) void qkv_gemm_kernel(
    const bf16_t* __restrict__ Hs, const bf16_t* __restrict__ Wqkv,
    const float* __restrict__ cosb, const float* __restrict__ sinb,
    const float* __restrict__ qw, const float* __restrict__ kw,
    bf16_t* __restrict__ q_ws, bf16_t* __restrict__ k_ws, bf16_t* __restrict__ v_ws)
{
  extern __shared__ char smem[];
  bf16_t* lds   = (bf16_t*)smem;                 // 4 staging tiles of 128*SA2
  float*  lds_c = (float*)smem;                  // 128 x SS (aliases staging after loop)

  const int t = threadIdx.x;
  const int lane = t & 31, wave = t >> 5;
  const int wm = wave >> 1, wn = wave & 1;
  const int head = blockIdx.x;                   // each N-tile == one head
  const int n0 = head * 128;
  const int m0 = blockIdx.y * 128;

  v8f acc[2][4];
#pragma unroll
  for (int i = 0; i < 2; ++i)
#pragma unroll
    for (int j = 0; j < 4; ++j) acc[i][j] = vzero8();

  const bf16_t* Ag = Hs   + (size_t)m0 * HID;
  const bf16_t* Bg = Wqkv + (size_t)n0 * HID;

  stage_tile64(Ag, lds, t);
  stage_tile64(Bg, lds + 2 * TILE_ELEMS, t);
  int cur = 0;

  for (int k0 = 0; k0 < HID; k0 += 64) {
    const bool has_next = (k0 + 64 < HID);
    if (has_next) {
      stage_tile64(Ag + k0 + 64, lds + (cur ^ 1) * TILE_ELEMS, t);
      stage_tile64(Bg + k0 + 64, lds + (2 + (cur ^ 1)) * TILE_ELEMS, t);
      async_wait8();                    // current tile done; next still in flight
    } else {
      async_wait0();
    }
    __syncthreads();

    bf16_t* sa = lds + cur * TILE_ELEMS;
    bf16_t* sb = lds + (2 + cur) * TILE_ELEMS;
#pragma unroll
    for (int kk = 0; kk < 64; kk += 32) {
      v16bf af[2], bfg[4];
#pragma unroll
      for (int tm = 0; tm < 2; ++tm)
        af[tm] = load_frag_a(sa + (wm * 32 + tm * 16) * SA2 + kk, SA2, lane);
#pragma unroll
      for (int tn = 0; tn < 4; ++tn)
        bfg[tn] = load_frag_b(sb + (wn * 64 + tn * 16) * SA2 + kk, SA2, lane);
#pragma unroll
      for (int tm = 0; tm < 2; ++tm)
#pragma unroll
        for (int tn = 0; tn < 4; ++tn)
          acc[tm][tn] = wmma_bf16(af[tm], bfg[tn], acc[tm][tn]);
    }
    __syncthreads();                    // reads done before this buffer is re-filled
    cur ^= 1;
  }

  // ---- epilogue: dump C tile to LDS, then per-row norm/rope/pack ----
  const int roff = (lane & 16) ? 8 : 0;
#pragma unroll
  for (int tm = 0; tm < 2; ++tm)
#pragma unroll
    for (int tn = 0; tn < 4; ++tn) {
      const int col = wn * 64 + tn * 16 + (lane & 15);
      const int rb  = wm * 32 + tm * 16 + roff;
#pragma unroll
      for (int r = 0; r < 8; ++r)
        lds_c[(rb + r) * SS + col] = acc[tm][tn][r];
    }
  __syncthreads();

  if (t < 128) {
    const int gm = m0 + t;
    const int bi = gm >> 11, si = gm & (SEQ - 1);
    float* crow = lds_c + t * SS;

    if (head < NH + NKV) {  // q or k head: RMSNorm + RoPE
      float ss = 0.f;
#pragma unroll 8
      for (int d = 0; d < HD; ++d) { float x = crow[d]; ss += x * x; }
      const float inv = rsqrtf(ss * (1.f / HD) + 1e-6f);
      const float* w = (head < NH) ? qw : kw;
#pragma unroll 8
      for (int d = 0; d < HD; ++d) crow[d] = crow[d] * inv * w[d];

      const float* cp = cosb + ((size_t)bi * SEQ + si) * 64;
      const float* sp = sinb + ((size_t)bi * SEQ + si) * 64;
      bf16_t* dst = (head < NH)
          ? q_ws + ((size_t)(bi * NH  + head)        * SEQ + si) * HD
          : k_ws + ((size_t)(bi * NKV + (head - NH)) * SEQ + si) * HD;
#pragma unroll
      for (int d = 0; d < 32; ++d) {
        float o1 = crow[d]      * cp[d]      - crow[d + 32] * sp[d];
        float o2 = crow[d + 32] * cp[d + 32] + crow[d]      * sp[d + 32];
        dst[d]      = (bf16_t)o1;
        dst[d + 32] = (bf16_t)o2;
      }
#pragma unroll 8
      for (int d = 64; d < HD; ++d) dst[d] = (bf16_t)crow[d];
    } else {                // v head: plain pack
      bf16_t* dst = v_ws + ((size_t)(bi * NKV + (head - NH - NKV)) * SEQ + si) * HD;
#pragma unroll 8
      for (int d = 0; d < HD; ++d) dst[d] = (bf16_t)crow[d];
    }
  }
}

// ---------------------------------------------------------------------------
// Kernel 2: causal flash attention, 128-row q tiles, 128-key k tiles.
// grid = (16 q-tiles, 16 heads, 2 batch), block = 256.
// ---------------------------------------------------------------------------
__global__ __launch_bounds__(256) void flash_attn_kernel(
    const bf16_t* __restrict__ q_ws, const bf16_t* __restrict__ k_ws,
    const bf16_t* __restrict__ v_ws, bf16_t* __restrict__ attn_ws)
{
  extern __shared__ char smem[];
  bf16_t* lds_q  = (bf16_t*)smem;                 // 128 x SQ
  bf16_t* lds_k  = lds_q  + 128 * SQ;             // 128 x SQ
  bf16_t* lds_vt = lds_k  + 128 * SQ;             // 128 x SQ  ([d][key])
  float*  lds_s  = (float*)(lds_vt + 128 * SQ);   // 128 x SS
  bf16_t* lds_p  = (bf16_t*)(lds_s + 128 * SS);   // 128 x SQ
  float*  lds_m  = (float*)(lds_p + 128 * SQ);    // 128
  float*  lds_l  = lds_m + 128;                   // 128
  float*  lds_al = lds_l + 128;                   // 128

  const int t = threadIdx.x, lane = t & 31, wave = t >> 5;
  const int wm = wave >> 1, wn = wave & 1;
  const int qt = blockIdx.x, head = blockIdx.y, bi = blockIdx.z;
  const int kvh = head >> 2;
  const int q0 = qt * 128;
  const int roff = (lane & 16) ? 8 : 0;

  // stage Q tile (async)
  {
    const int row = t >> 1, cb = (t & 1) * 64;
    const bf16_t* s = q_ws + ((size_t)(bi * NH + head) * SEQ + q0 + row) * HD + cb;
    bf16_t* d = lds_q + row * SQ + cb;
#pragma unroll
    for (int i = 0; i < 8; ++i) async_copy_b128(d + i * 8, s + i * 8);
  }
  if (t < 128) { lds_m[t] = -3.0e38f; lds_l[t] = 0.f; }

  v8f acco[2][4];
#pragma unroll
  for (int i = 0; i < 2; ++i)
#pragma unroll
    for (int j = 0; j < 4; ++j) acco[i][j] = vzero8();

  for (int kt = 0; kt <= qt; ++kt) {
    const int kb = kt * 128;
    __syncthreads();
    {  // stage K tile ([key][d] == [N][K]) (async)
      const int row = t >> 1, cb = (t & 1) * 64;
      const bf16_t* s = k_ws + ((size_t)(bi * NKV + kvh) * SEQ + kb + row) * HD + cb;
      bf16_t* d = lds_k + row * SQ + cb;
#pragma unroll
      for (int i = 0; i < 8; ++i) async_copy_b128(d + i * 8, s + i * 8);
    }
    {  // stage V tile transposed -> [d][key] (VALU path, needs transpose)
      const int key = t >> 1, db = (t & 1) * 64;
      const v8bf* s = (const v8bf*)(v_ws + ((size_t)(bi * NKV + kvh) * SEQ + kb + key) * HD + db);
#pragma unroll
      for (int i = 0; i < 8; ++i) {
        v8bf x = s[i];
#pragma unroll
        for (int j = 0; j < 8; ++j) lds_vt[(db + i * 8 + j) * SQ + key] = x[j];
      }
    }
    async_wait0();
    __syncthreads();

    // S = Q K^T  (K-dim = head_dim 128, 4 WMMA k-steps)
    v8f accs[2][4];
#pragma unroll
    for (int i = 0; i < 2; ++i)
#pragma unroll
      for (int j = 0; j < 4; ++j) accs[i][j] = vzero8();
#pragma unroll
    for (int kk = 0; kk < HD; kk += 32) {
      v16bf af[2], bfg[4];
#pragma unroll
      for (int tm = 0; tm < 2; ++tm)
        af[tm] = load_frag_a(lds_q + (wm * 32 + tm * 16) * SQ + kk, SQ, lane);
#pragma unroll
      for (int tn = 0; tn < 4; ++tn)
        bfg[tn] = load_frag_b(lds_k + (wn * 64 + tn * 16) * SQ + kk, SQ, lane);
#pragma unroll
      for (int tm = 0; tm < 2; ++tm)
#pragma unroll
        for (int tn = 0; tn < 4; ++tn)
          accs[tm][tn] = wmma_bf16(af[tm], bfg[tn], accs[tm][tn]);
    }
    // spill S to LDS for row-wise softmax
#pragma unroll
    for (int tm = 0; tm < 2; ++tm)
#pragma unroll
      for (int tn = 0; tn < 4; ++tn) {
        const int col = wn * 64 + tn * 16 + (lane & 15);
        const int rb  = wm * 32 + tm * 16 + roff;
#pragma unroll
        for (int r = 0; r < 8; ++r)
          lds_s[(rb + r) * SS + col] = accs[tm][tn][r];
      }
    __syncthreads();

    if (t < 128) {  // online softmax, one row per thread
      const int row = t, qpos = q0 + row;
      const float sc = 0.08838834764831845f;     // 1/sqrt(128)
      float* srow = lds_s + row * SS;
      const int lim = qpos - kb;
      const int cmax = (lim >= 127) ? 128 : (lim + 1);
      const float mo = lds_m[row];
      float mx = mo;
      for (int c = 0; c < cmax; ++c) mx = fmaxf(mx, srow[c] * sc);
      const float alpha = __expf(mo - mx);
      float sum = 0.f;
      bf16_t* prow = lds_p + row * SQ;
      for (int c = 0; c < cmax; ++c) {
        const float e = __expf(srow[c] * sc - mx);
        sum += e;
        prow[c] = (bf16_t)e;
      }
      for (int c = cmax; c < 128; ++c) prow[c] = (bf16_t)0.f;
      lds_m[row]  = mx;
      lds_l[row]  = lds_l[row] * alpha + sum;
      lds_al[row] = alpha;
    }
    __syncthreads();

    // rescale O, then O += P * V  (K-dim = keys 128)
#pragma unroll
    for (int tm = 0; tm < 2; ++tm)
#pragma unroll
      for (int tn = 0; tn < 4; ++tn) {
        const int rb = wm * 32 + tm * 16 + roff;
#pragma unroll
        for (int r = 0; r < 8; ++r)
          acco[tm][tn][r] *= lds_al[rb + r];
      }
#pragma unroll
    for (int kk = 0; kk < 128; kk += 32) {
      v16bf af[2], bfg[4];
#pragma unroll
      for (int tm = 0; tm < 2; ++tm)
        af[tm] = load_frag_a(lds_p + (wm * 32 + tm * 16) * SQ + kk, SQ, lane);
#pragma unroll
      for (int tn = 0; tn < 4; ++tn)
        bfg[tn] = load_frag_b(lds_vt + (wn * 64 + tn * 16) * SQ + kk, SQ, lane);
#pragma unroll
      for (int tm = 0; tm < 2; ++tm)
#pragma unroll
        for (int tn = 0; tn < 4; ++tn)
          acco[tm][tn] = wmma_bf16(af[tm], bfg[tn], acco[tm][tn]);
    }
  }

  // finalize: O /= l, pack bf16 into [b][s][h*128]
#pragma unroll
  for (int tm = 0; tm < 2; ++tm)
#pragma unroll
    for (int tn = 0; tn < 4; ++tn) {
      const int col = wn * 64 + tn * 16 + (lane & 15);
      const int rb  = wm * 32 + tm * 16 + roff;
#pragma unroll
      for (int r = 0; r < 8; ++r) {
        const int row = rb + r;
        const float invl = 1.f / lds_l[row];
        attn_ws[((size_t)bi * SEQ + q0 + row) * (NH * HD) + head * HD + col] =
            (bf16_t)(acco[tm][tn][r] * invl);
      }
    }
}

// ---------------------------------------------------------------------------
// Kernel 3: dense projection (out = attn * W_dense^T), async double-buffered.
// grid = (16 N-tiles, 32 M-tiles), block = 256.
// ---------------------------------------------------------------------------
__global__ __launch_bounds__(256) void dense_gemm_kernel(
    const bf16_t* __restrict__ A, const bf16_t* __restrict__ Wd,
    float* __restrict__ out)
{
  extern __shared__ char smem[];
  bf16_t* lds = (bf16_t*)smem;                   // 4 staging tiles of 128*SA2

  const int t = threadIdx.x, lane = t & 31, wave = t >> 5;
  const int wm = wave >> 1, wn = wave & 1;
  const int n0 = blockIdx.x * 128, m0 = blockIdx.y * 128;
  const int roff = (lane & 16) ? 8 : 0;

  v8f acc[2][4];
#pragma unroll
  for (int i = 0; i < 2; ++i)
#pragma unroll
    for (int j = 0; j < 4; ++j) acc[i][j] = vzero8();

  const bf16_t* Ag = A  + (size_t)m0 * HID;
  const bf16_t* Bg = Wd + (size_t)n0 * HID;

  stage_tile64(Ag, lds, t);
  stage_tile64(Bg, lds + 2 * TILE_ELEMS, t);
  int cur = 0;

  for (int k0 = 0; k0 < HID; k0 += 64) {
    const bool has_next = (k0 + 64 < HID);
    if (has_next) {
      stage_tile64(Ag + k0 + 64, lds + (cur ^ 1) * TILE_ELEMS, t);
      stage_tile64(Bg + k0 + 64, lds + (2 + (cur ^ 1)) * TILE_ELEMS, t);
      async_wait8();
    } else {
      async_wait0();
    }
    __syncthreads();

    bf16_t* sa = lds + cur * TILE_ELEMS;
    bf16_t* sb = lds + (2 + cur) * TILE_ELEMS;
#pragma unroll
    for (int kk = 0; kk < 64; kk += 32) {
      v16bf af[2], bfg[4];
#pragma unroll
      for (int tm = 0; tm < 2; ++tm)
        af[tm] = load_frag_a(sa + (wm * 32 + tm * 16) * SA2 + kk, SA2, lane);
#pragma unroll
      for (int tn = 0; tn < 4; ++tn)
        bfg[tn] = load_frag_b(sb + (wn * 64 + tn * 16) * SA2 + kk, SA2, lane);
#pragma unroll
      for (int tm = 0; tm < 2; ++tm)
#pragma unroll
        for (int tn = 0; tn < 4; ++tn)
          acc[tm][tn] = wmma_bf16(af[tm], bfg[tn], acc[tm][tn]);
    }
    __syncthreads();
    cur ^= 1;
  }

#pragma unroll
  for (int tm = 0; tm < 2; ++tm)
#pragma unroll
    for (int tn = 0; tn < 4; ++tn) {
      const int col = n0 + wn * 64 + tn * 16 + (lane & 15);
      const int rb  = m0 + wm * 32 + tm * 16 + roff;
#pragma unroll
      for (int r = 0; r < 8; ++r)
        out[(size_t)(rb + r) * HID + col] = acc[tm][tn][r];
    }
}

// ---------------------------------------------------------------------------
extern "C" void kernel_launch(void* const* d_in, const int* in_sizes, int n_in,
                              void* d_out, int out_size, void* d_ws, size_t ws_size,
                              hipStream_t stream) {
  (void)in_sizes; (void)n_in; (void)out_size; (void)ws_size;
  const float* Hs   = (const float*)d_in[0];
  // d_in[1] = attention_mask: exactly causal -> applied analytically, not read
  const float* cosb = (const float*)d_in[2];
  const float* sinb = (const float*)d_in[3];
  const float* Wqkv = (const float*)d_in[4];
  const float* qw   = (const float*)d_in[5];
  const float* kw   = (const float*)d_in[6];
  const float* Wd   = (const float*)d_in[7];
  float* out = (float*)d_out;

  bf16_t* q_ws   = (bf16_t*)d_ws;                             // [2][16][2048][128]
  bf16_t* k_ws   = q_ws   + (size_t)2 * NH  * SEQ * HD;       // [2][4][2048][128]
  bf16_t* v_ws   = k_ws   + (size_t)2 * NKV * SEQ * HD;       // [2][4][2048][128]
  bf16_t* attn   = v_ws   + (size_t)2 * NKV * SEQ * HD;       // [4096][2048]
  bf16_t* h_bf   = attn   + (size_t)2 * SEQ * NH * HD;        // [4096][2048]
  bf16_t* wqkv_bf= h_bf   + (size_t)2 * SEQ * HID;            // [3072][2048]
  bf16_t* wd_bf  = wqkv_bf+ (size_t)(NH + 2 * NKV) * HD * HID;// [2048][2048]

  const int n_h  = 2 * SEQ * HID;                  // 8388608
  const int n_wq = (NH + 2 * NKV) * HD * HID;      // 6291456
  const int n_wd = HID * HID;                      // 4194304

  cvt_bf16_kernel<<<n_h  / 2048, 256, 0, stream>>>(Hs,   h_bf,    n_h);
  cvt_bf16_kernel<<<n_wq / 2048, 256, 0, stream>>>(Wqkv, wqkv_bf, n_wq);
  cvt_bf16_kernel<<<n_wd / 2048, 256, 0, stream>>>(Wd,   wd_bf,   n_wd);

  const size_t smem_gemm = (size_t)4 * TILE_ELEMS * 2;                // 73728 (>= 128*SS*4)
  const size_t smem_attn = (size_t)3 * 128 * SQ * 2 + 128 * SS * 4
                         + 128 * SQ * 2 + 3 * 128 * 4;                // 208384

  qkv_gemm_kernel<<<dim3(NH + 2 * NKV, 32), 256, smem_gemm, stream>>>(
      h_bf, wqkv_bf, cosb, sinb, qw, kw, q_ws, k_ws, v_ws);
  flash_attn_kernel<<<dim3(SEQ / 128, NH, 2), 256, smem_attn, stream>>>(
      q_ws, k_ws, v_ws, attn);
  dense_gemm_kernel<<<dim3(HID / 128, 32), 256, smem_gemm, stream>>>(
      attn, wd_bf, out);
}
